// SimpleCodebook_39822936768746
// MI455X (gfx1250) — compile-verified, matrix operand order checked by reading, not run on hardware
//
#include <hip/hip_runtime.h>
#include <stdint.h>

#define DIM   512
#define NVEC  8192   // 8 * 1024 query vectors
#define KCODE 8192   // codebook size

// LDS tile geometry: 64 code rows; each row block = [1024B hi][1024B lo][16B pad]
// stride 2064B = 516 dwords == 4 (mod 64 banks) -> breaks row-stride conflicts,
// and keeps every hi/lo/ns/k access a 16-bit DS immediate offset from 2 bases.
#define ROW_STRIDE 2064
#define NS_STRIDE  (16 * ROW_STRIDE)          // 33024 B per 16-code group
#define BUF_BYTES  (64 * ROW_STRIDE)          // one tile (hi+lo interleaved): 132096 B
#define LDS_TOTAL  (2 * BUF_BYTES)            // double buffered:              264192 B

typedef __attribute__((ext_vector_type(16))) __bf16 v16bf;
typedef __attribute__((ext_vector_type(8)))  float  v8f;

// ---------- bf16 split helpers (round-to-nearest-even, bit-level) ----------
__device__ __forceinline__ uint16_t f32_to_bf16_rne(float f) {
    union { float f; uint32_t u; } c; c.f = f;
    uint32_t u = c.u;
    uint32_t r = u + 0x7FFFu + ((u >> 16) & 1u);
    return (uint16_t)(r >> 16);
}
__device__ __forceinline__ float bf16_bits_to_f32(uint16_t h) {
    union { uint32_t u; float f; } c; c.u = ((uint32_t)h) << 16;
    return c.f;
}

// ---------- kernel 1: split X into bf16 hi/lo ----------
__global__ void vq_split_x(const float* __restrict__ x,
                           uint16_t* __restrict__ xh, uint16_t* __restrict__ xl, int n) {
    int i = blockIdx.x * blockDim.x + threadIdx.x;
    if (i >= n) return;
    float v = x[i];
    uint16_t h = f32_to_bf16_rne(v);
    xh[i] = h;
    xl[i] = f32_to_bf16_rne(v - bf16_bits_to_f32(h));
}

// ---------- kernel 2: split E into bf16 hi/lo + compute ||e||^2 per row ----------
__global__ void vq_split_e(const float* __restrict__ e,
                           uint16_t* __restrict__ eh, uint16_t* __restrict__ el,
                           float* __restrict__ esq) {
    __shared__ float red[256];
    const int row = blockIdx.x;
    const int t   = threadIdx.x;
    float ssum = 0.f;
    for (int j = t; j < DIM; j += 256) {
        float v = e[(size_t)row * DIM + j];
        uint16_t h = f32_to_bf16_rne(v);
        eh[(size_t)row * DIM + j] = h;
        el[(size_t)row * DIM + j] = f32_to_bf16_rne(v - bf16_bits_to_f32(h));
        ssum += v * v;
    }
    red[t] = ssum;
    __syncthreads();
    for (int s = 128; s > 0; s >>= 1) {
        if (t < s) red[t] += red[t + s];
        __syncthreads();
    }
    if (t == 0) esq[row] = red[0];
}

// ---------- async stage: one 64-code x 512 (hi+lo) tile -> LDS buffer ----------
// 256 threads x 16 iters x 16B x 2 arrays = 128 KB; 32 async ops per thread.
__device__ __forceinline__ void stage_tile(char* smem, uint32_t bufBase,
                                           const uint16_t* __restrict__ Eh,
                                           const uint16_t* __restrict__ El,
                                           int row0, int tid) {
#pragma unroll
    for (int i = 0; i < 16; ++i) {
        const int c   = tid + i * 256;      // chunk 0..4095
        const int row = c >> 6;             // 0..63
        const int col = c & 63;             // 16B chunk within row
        uint32_t lds_hi = (uint32_t)(uintptr_t)(smem + bufBase + row * ROW_STRIDE + col * 16);
        uint32_t lds_lo = lds_hi + 1024;
        uint64_t g_hi = (uint64_t)(uintptr_t)(Eh + (size_t)(row0 + row) * DIM + col * 8);
        uint64_t g_lo = (uint64_t)(uintptr_t)(El + (size_t)(row0 + row) * DIM + col * 8);
        asm volatile("global_load_async_to_lds_b128 %0, %1, off"
                     :: "v"(lds_hi), "v"(g_hi) : "memory");
        asm volatile("global_load_async_to_lds_b128 %0, %1, off"
                     :: "v"(lds_lo), "v"(g_lo) : "memory");
    }
}

// ---------- kernel 3: bf16x3 WMMA GEMM + fused argmax ----------
// grid: (NVEC/128, KCODE/512); block 256 = 8 waves.
// Each wave: 16 rows x 64 codes (4 accumulators); all waves share the LDS B tile.
__global__ __launch_bounds__(256)
void vq_gemm_argmax(const uint16_t* __restrict__ Xh, const uint16_t* __restrict__ Xl,
                    const uint16_t* __restrict__ Eh, const uint16_t* __restrict__ El,
                    const float* __restrict__ esq,
                    float* __restrict__ pscore, int* __restrict__ pidx) {
    extern __shared__ char smem[];

    const int tid  = threadIdx.x;
    const int wave = tid >> 5;          // 0..7 = M sub-tile
    const int lane = tid & 31;
    const int half = lane >> 4;         // k-chunk half
    const int lm   = lane & 15;

    const int rowBase   = blockIdx.x * 128;
    const int stripBase = blockIdx.y * 512;

    // A operand pointers: lane lm supplies row (rowBase + wave*16 + lm)
    const int arow = rowBase + wave * 16 + lm;
    const uint16_t* aHi = Xh + (size_t)arow * DIM + half * 16;
    const uint16_t* aLo = Xl + (size_t)arow * DIM + half * 16;

    // Per-lane LDS base for B reads (all further addressing is immediate offsets)
    const char* bpLane = smem + lm * ROW_STRIDE + half * 32;

    float best[8];
    int   bidx[8];
#pragma unroll
    for (int v = 0; v < 8; ++v) { best[v] = -__builtin_huge_valf(); bidx[v] = 0; }

    // prime double buffer with tile 0
    stage_tile(smem, 0u, Eh, El, stripBase, tid);

    for (int t = 0; t < 8; ++t) {
        const uint32_t bufBase = (uint32_t)(t & 1) * BUF_BYTES;
        if (t < 7) {
            stage_tile(smem, (uint32_t)((t + 1) & 1) * BUF_BYTES, Eh, El,
                       stripBase + (t + 1) * 64, tid);
            // async loads complete in order: <=32 outstanding => tile t fully in LDS
            asm volatile("s_wait_asynccnt 0x20" ::: "memory");
        } else {
            asm volatile("s_wait_asynccnt 0x0" ::: "memory");
        }
        __syncthreads();

        const int tileBase = stripBase + t * 64;
        const char* bp0 = bpLane + bufBase;          // ns 0 / 1 base
        const char* bp2 = bp0 + 2 * NS_STRIDE;       // ns 2 / 3 base

        v8f acc[4];
#pragma unroll
        for (int ns = 0; ns < 4; ++ns) acc[ns] = (v8f){0.f,0.f,0.f,0.f,0.f,0.f,0.f,0.f};

#pragma unroll 4
        for (int ks = 0; ks < 16; ++ks) {
            const int kb = ks * 64;                   // byte offset along k
            v16bf ah = *(const v16bf*)(aHi + ks * 32);  // global (L2-resident)
            v16bf al = *(const v16bf*)(aLo + ks * 32);

            // hoist all B operands: 16 ds_load_b128, pure immediate offsets
            v16bf bh[4], bl[4];
#pragma unroll
            for (int ns = 0; ns < 4; ++ns) {
                const char* bp = (ns < 2 ? bp0 : bp2) + (ns & 1) * NS_STRIDE + kb;
                bh[ns] = *(const v16bf*)(bp);
                bl[ns] = *(const v16bf*)(bp + 1024);
            }
#pragma unroll
            for (int ns = 0; ns < 4; ++ns) {
                acc[ns] = __builtin_amdgcn_wmma_f32_16x16x32_bf16(false, ah, false, bh[ns], (short)0, acc[ns], false, false);
                acc[ns] = __builtin_amdgcn_wmma_f32_16x16x32_bf16(false, ah, false, bl[ns], (short)0, acc[ns], false, false);
                acc[ns] = __builtin_amdgcn_wmma_f32_16x16x32_bf16(false, al, false, bh[ns], (short)0, acc[ns], false, false);
            }
        }

        // score = 2*x.e - ||e||^2 ; per-lane running argmax (lane's column = code)
#pragma unroll
        for (int ns = 0; ns < 4; ++ns) {
            const int code = tileBase + ns * 16 + lm;
            const float es = esq[code];
#pragma unroll
            for (int v = 0; v < 8; ++v) {
                float s = 2.0f * acc[ns][v] - es;
                if (s > best[v]) { best[v] = s; bidx[v] = code; }
            }
        }
        __syncthreads();   // done reading bufBase before it is re-staged
    }

    // cross-lane argmax within each 16-lane half (C cols live across lanes)
#pragma unroll
    for (int m = 1; m < 16; m <<= 1) {
#pragma unroll
        for (int v = 0; v < 8; ++v) {
            float os = __shfl_xor(best[v], m, 32);
            int   oi = __shfl_xor(bidx[v], m, 32);
            if (os > best[v] || (os == best[v] && oi < bidx[v])) { best[v] = os; bidx[v] = oi; }
        }
    }

    if (lm == 0) {
        const int slot = blockIdx.y;          // 16 strip candidates per row
#pragma unroll
        for (int v = 0; v < 8; ++v) {
            const int row = rowBase + wave * 16 + v + half * 8;
            pscore[(size_t)row * 16 + slot] = best[v];
            pidx  [(size_t)row * 16 + slot] = bidx[v];
        }
    }
}

// ---------- kernel 4: reduce 16 strip candidates per row, gather code row ----------
__global__ void vq_finalize(const float* __restrict__ pscore, const int* __restrict__ pidx,
                            const float* __restrict__ E, float* __restrict__ out) {
    const int row = blockIdx.x;
    float bs = -__builtin_huge_valf();
    int   bi = 0;
#pragma unroll
    for (int s = 0; s < 16; ++s) {
        float v = pscore[(size_t)row * 16 + s];
        int   i = pidx  [(size_t)row * 16 + s];
        if (v > bs || (v == bs && i < bi)) { bs = v; bi = i; }
    }
    const float* src = E + (size_t)bi * DIM;
    for (int j = threadIdx.x; j < DIM; j += blockDim.x)
        out[(size_t)row * DIM + j] = src[j];
    if (threadIdx.x == 0)
        out[(size_t)NVEC * DIM + row] = (float)bi;   // embed_ind after quantize block
}

extern "C" void kernel_launch(void* const* d_in, const int* in_sizes, int n_in,
                              void* d_out, int out_size, void* d_ws, size_t ws_size,
                              hipStream_t stream) {
    (void)in_sizes; (void)n_in; (void)out_size; (void)ws_size;
    const float* x = (const float*)d_in[0];   // [8,1024,512]
    const float* e = (const float*)d_in[1];   // [8192,512]

    char* ws = (char*)d_ws;
    size_t off = 0;
    uint16_t* Xh = (uint16_t*)(ws + off); off += (size_t)NVEC  * DIM * sizeof(uint16_t);
    uint16_t* Xl = (uint16_t*)(ws + off); off += (size_t)NVEC  * DIM * sizeof(uint16_t);
    uint16_t* Eh = (uint16_t*)(ws + off); off += (size_t)KCODE * DIM * sizeof(uint16_t);
    uint16_t* El = (uint16_t*)(ws + off); off += (size_t)KCODE * DIM * sizeof(uint16_t);
    float* esq    = (float*)(ws + off);   off += (size_t)KCODE * sizeof(float);
    float* pscore = (float*)(ws + off);   off += (size_t)NVEC * 16 * sizeof(float);
    int*   pidx   = (int*)(ws + off);     off += (size_t)NVEC * 16 * sizeof(int);

    vq_split_x<<<(NVEC * DIM) / 256, 256, 0, stream>>>(x, Xh, Xl, NVEC * DIM);
    vq_split_e<<<KCODE, 256, 0, stream>>>(e, Eh, El, esq);

    dim3 grid(NVEC / 128, KCODE / 512);   // 64 x 16 workgroups
    vq_gemm_argmax<<<grid, 256, LDS_TOTAL, stream>>>(Xh, Xl, Eh, El, esq, pscore, pidx);

    vq_finalize<<<NVEC, 128, 0, stream>>>(pscore, pidx, e, (float*)d_out);
}
